// Attention_Generator_16741782520573
// MI455X (gfx1250) — matrix operand order repaired and checked by reading, not compiled, not run on hardware
//
#include <hip/hip_runtime.h>
#include <math.h>

// Problem constants (match reference file)
#define BB   64
#define TT   21
#define LL   36
#define EE   512
#define HH   512
#define FF   1024
#define VV   20000
#define ATT_IN  (HH + EE + FF)   // 2048
#define LANG_IN (FF + HH)        // 1536
#define G3H     (3 * HH)         // 1536
#define TSTEPS  (TT - 1)         // 20

typedef __attribute__((ext_vector_type(16))) __bf16 v16bf;
typedef __attribute__((ext_vector_type(8)))  float  v8f;
typedef __attribute__((ext_vector_type(4)))  float  v4f;
typedef __attribute__((ext_vector_type(2)))  unsigned int v2u;

__device__ __forceinline__ unsigned int bf16bits(float f) {
    union { float f; unsigned u; } x; x.f = f;
    unsigned u = x.u + 0x7FFFu + ((x.u >> 16) & 1u);   // round-to-nearest-even
    return u >> 16;
}
// pack two fp32 -> one dword of two bf16 (lo = first)
__device__ __forceinline__ unsigned int pack2bf(float a, float b) {
    return bf16bits(a) | (bf16bits(b) << 16);
}

__device__ __forceinline__ float sigmoidf(float x) { return 1.0f / (1.0f + expf(-x)); }

// ---------------------------------------------------------------------------
// WMMA GEMM:  C[M,N] = A[M,K] @ W[N,K]^T + bias[N]
// A fp32 row-major (lda), W fp32 row-major (ldw) -- PyTorch (out,in) layout.
// fp32 loaded as float4, converted to bf16, staged in LDS; fp32 accumulate
// via v_wmma_f32_16x16x32_bf16.
// Block: 128 threads = 4 wave32. Block tile 64(M) x 32(N); each wave owns a
// 16x32 strip -> two accumulators, A fragment reused across both WMMAs.
// Requires M%64==0, N%32==0, K%32==0 (all call sites satisfy this); all
// control flow is uniform so EXEC is all-ones at every WMMA.
// ---------------------------------------------------------------------------
__global__ void __launch_bounds__(128)
gemm_bf16_wmma(const float* __restrict__ A, int lda,
               const float* __restrict__ W, int ldw,
               const float* __restrict__ bias,
               float* __restrict__ C, int ldc,
               int M, int N, int K)
{
    // LDS tiles in dword units: row = 32 bf16 = 16 dwords
    __shared__ unsigned int sA[64 * 16];   // [m][k/2]
    __shared__ unsigned int sB[32 * 16];   // [n][k/2]

    const int tid  = threadIdx.x;     // 0..127
    const int wave = tid >> 5;        // 0..3  -> row group of 16
    const int lane = tid & 31;
    const int half = lane >> 4;       // 0/1
    const int l16  = lane & 15;
    const int mBlk = blockIdx.y * 64;
    const int nBlk = blockIdx.x * 32;

    v8f acc0 = {};
    v8f acc1 = {};
    const int mrow = wave * 16 + l16;

    for (int k0 = 0; k0 < K; k0 += 32) {
        __syncthreads();
        // ---- stage A tile (64 rows x 32 k) : 4 fixed iterations, float4 loads
        #pragma unroll
        for (int it = 0; it < 4; ++it) {
            int iu = (it * 128 + tid) * 2;          // dword index, covers 64*16
            int m  = iu >> 4, ku = iu & 15;         // element k = ku*2
            const v4f f = *(const v4f*)&A[(size_t)(mBlk + m) * lda + (k0 + ku * 2)];
            v2u p; p.x = pack2bf(f.x, f.y); p.y = pack2bf(f.z, f.w);
            *(v2u*)&sA[iu] = p;
        }
        // ---- stage B tile (32 n-rows x 32 k) : 2 fixed iterations
        #pragma unroll
        for (int it = 0; it < 2; ++it) {
            int iu = (it * 128 + tid) * 2;          // dword index, covers 32*16
            int n  = iu >> 4, ku = iu & 15;
            const v4f f = *(const v4f*)&W[(size_t)(nBlk + n) * ldw + (k0 + ku * 2)];
            v2u p; p.x = pack2bf(f.x, f.y); p.y = pack2bf(f.z, f.w);
            *(v2u*)&sB[iu] = p;
        }
        // uniform-guard prefetch of next K tile (weights stream from L2)
        if (k0 + 32 < K) {
            int iu = tid * 2;
            int m = iu >> 4, ku = iu & 15;
            __builtin_prefetch(&A[(size_t)(mBlk + m) * lda + (k0 + 32 + ku * 2)], 0, 1);
            int n = (tid >> 4) & 31;
            __builtin_prefetch(&W[(size_t)(nBlk + n) * ldw + (k0 + 32 + ((tid & 15) * 2))], 0, 1);
        }
        __syncthreads();

        // ---- build fragments per wave32 VGPR layouts (CDNA5 ISA)
        union { v16bf v; unsigned u[8]; } fa, fb0, fb1;
        #pragma unroll
        for (int j = 0; j < 8; ++j) {
            // A 16x32 bf16: lanes0-15 dword j<4 -> K-pair j ; j>=4 -> pair j+4
            //               lanes16-31: +4 dwords (K base +8)
            int kaU = ((j < 4) ? j : (j + 4)) + 4 * half;
            fa.u[j] = sA[mrow * 16 + kaU];
            // B 32x16 bf16: col = l16 (+16 for second tile), same K order as A
            //               lanes16-31: +4 dwords (K base +8)
            int kbU = ((j < 4) ? j : (j + 4)) + 4 * half;
            fb0.u[j] = sB[(l16)      * 16 + kbU];
            fb1.u[j] = sB[(16 + l16) * 16 + kbU];
        }
        acc0 = __builtin_amdgcn_wmma_f32_16x16x32_bf16(
                   false, fa.v, false, fb0.v, (short)0, acc0, false, false);
        acc1 = __builtin_amdgcn_wmma_f32_16x16x32_bf16(
                   false, fa.v, false, fb1.v, (short)0, acc1, false, false);
    }

    // C/D layout: dword r -> row = r + 8*half, col = l16 (per 16x16 tile)
    const float bv0 = bias ? bias[nBlk + l16]      : 0.0f;
    const float bv1 = bias ? bias[nBlk + 16 + l16] : 0.0f;
    #pragma unroll
    for (int r = 0; r < 8; ++r) {
        int row = mBlk + wave * 16 + 8 * half + r;
        C[(size_t)row * ldc + nBlk + l16]      = acc0[r] + bv0;
        C[(size_t)row * ldc + nBlk + 16 + l16] = acc1[r] + bv1;
    }
}

// ---------------------------------------------------------------------------
// Small helper kernels (elementwise / tiny reductions)
// ---------------------------------------------------------------------------
__global__ void zero_kernel(float* p, int n) {
    int i = blockIdx.x * blockDim.x + threadIdx.x;
    if (i < n) p[i] = 0.0f;
}

__global__ void vmean_kernel(const float* __restrict__ fv, float* __restrict__ vmean) {
    int i = blockIdx.x * blockDim.x + threadIdx.x;          // over B*F
    if (i >= BB * FF) return;
    int b = i / FF, f = i % FF;
    float s = 0.0f;
    for (int l = 0; l < LL; ++l) s += fv[((size_t)b * LL + l) * FF + f];
    vmean[i] = s * (1.0f / LL);
}

// x_att = concat([h2, x_t, v_mean])  (B x 2048)
__global__ void pack_att_kernel(const float* __restrict__ h2,
                                const float* __restrict__ gt,   // (B,T,E)
                                const float* __restrict__ vmean,
                                float* __restrict__ x, int t) {
    int i = blockIdx.x * blockDim.x + threadIdx.x;
    if (i >= BB * ATT_IN) return;
    int b = i / ATT_IN, c = i % ATT_IN;
    float v;
    if (c < HH)            v = h2[b * HH + c];
    else if (c < HH + EE)  v = gt[((size_t)b * TT + t) * EE + (c - HH)];
    else                   v = vmean[b * FF + (c - HH - EE)];
    x[i] = v;
}

// x_lang = concat([mix, h1])  (B x 1536)
__global__ void pack_lang_kernel(const float* __restrict__ mix,
                                 const float* __restrict__ h1,
                                 float* __restrict__ x) {
    int i = blockIdx.x * blockDim.x + threadIdx.x;
    if (i >= BB * LANG_IN) return;
    int b = i / LANG_IN, c = i % LANG_IN;
    x[i] = (c < FF) ? mix[b * FF + c] : h1[b * HH + (c - FF)];
}

// GRU gate fusion: h <- (1-z)*n + z*h  (in-place on h)
__global__ void gru_gate_kernel(const float* __restrict__ gi,
                                const float* __restrict__ gh,
                                float* __restrict__ h) {
    int i = blockIdx.x * blockDim.x + threadIdx.x;          // over B*H
    if (i >= BB * HH) return;
    int b = i / HH, j = i % HH;
    const float* gib = gi + (size_t)b * G3H;
    const float* ghb = gh + (size_t)b * G3H;
    float r = sigmoidf(gib[j]          + ghb[j]);
    float z = sigmoidf(gib[HH + j]     + ghb[HH + j]);
    float n = tanhf   (gib[2 * HH + j] + r * ghb[2 * HH + j]);
    h[i] = (1.0f - z) * n + z * h[i];
}

// attention scores + softmax over L=36: one block per batch row
__global__ void att_score_kernel(const float* __restrict__ v_w,   // (B*L,128)
                                 const float* __restrict__ h_w,   // (B,128)
                                 const float* __restrict__ fc3_w, // (128)
                                 const float* __restrict__ fc3_b, // (1)
                                 float* __restrict__ attw) {      // (B,L)
    __shared__ float s[LL];
    int b = blockIdx.x;
    int l = threadIdx.x;
    if (l < LL) {
        const float* vw = v_w + ((size_t)b * LL + l) * 128;
        const float* hw = h_w + (size_t)b * 128;
        float acc = fc3_b[0];
        #pragma unroll 4
        for (int j = 0; j < 128; ++j) acc += fc3_w[j] * tanhf(vw[j] + hw[j]);
        s[l] = acc;
    }
    __syncthreads();
    if (threadIdx.x == 0) {
        float mx = s[0];
        for (int k = 1; k < LL; ++k) mx = fmaxf(mx, s[k]);
        float sum = 0.0f;
        for (int k = 0; k < LL; ++k) { s[k] = expf(s[k] - mx); sum += s[k]; }
        float inv = 1.0f / sum;
        for (int k = 0; k < LL; ++k) attw[b * LL + k] = s[k] * inv;
    }
}

// mix[b,f] = sum_l a[b,l] * fv[b,l,f]
__global__ void mix_kernel(const float* __restrict__ attw,
                           const float* __restrict__ fv,
                           float* __restrict__ mix) {
    int i = blockIdx.x * blockDim.x + threadIdx.x;          // over B*F
    if (i >= BB * FF) return;
    int b = i / FF, f = i % FF;
    float s = 0.0f;
    for (int l = 0; l < LL; ++l)
        s += attw[b * LL + l] * fv[((size_t)b * LL + l) * FF + f];
    mix[i] = s;
}

// ---------------------------------------------------------------------------
// Host launcher
// ---------------------------------------------------------------------------
static inline void launch_gemm(const float* A, int lda, const float* W, int ldw,
                               const float* bias, float* C, int ldc,
                               int M, int N, int K, hipStream_t stream) {
    dim3 grid(N / 32, M / 64);
    gemm_bf16_wmma<<<grid, 128, 0, stream>>>(A, lda, W, ldw, bias, C, ldc, M, N, K);
}

extern "C" void kernel_launch(void* const* d_in, const int* in_sizes, int n_in,
                              void* d_out, int out_size, void* d_ws, size_t ws_size,
                              hipStream_t stream) {
    (void)in_sizes; (void)n_in; (void)out_size; (void)ws_size;

    const float* fv        = (const float*)d_in[0];   // (B,L,F)
    const float* gt        = (const float*)d_in[2];   // (B,T,E)
    const float* att_w_ih  = (const float*)d_in[3];   // (3H, 2048)
    const float* att_w_hh  = (const float*)d_in[4];   // (3H, 512)
    const float* att_b_ih  = (const float*)d_in[5];
    const float* att_b_hh  = (const float*)d_in[6];
    const float* lang_w_ih = (const float*)d_in[7];   // (3H, 1536)
    const float* lang_w_hh = (const float*)d_in[8];   // (3H, 512)
    const float* lang_b_ih = (const float*)d_in[9];
    const float* lang_b_hh = (const float*)d_in[10];
    const float* fc1_w     = (const float*)d_in[11];  // (128, 1024)
    const float* fc1_b     = (const float*)d_in[12];
    const float* fc2_w     = (const float*)d_in[13];  // (128, 512)
    const float* fc2_b     = (const float*)d_in[14];
    const float* fc3_w     = (const float*)d_in[15];  // (1, 128)
    const float* fc3_b     = (const float*)d_in[16];
    const float* dense_w   = (const float*)d_in[17];  // (VOC, 512)
    const float* dense_b   = (const float*)d_in[18];
    float* out             = (float*)d_out;           // (B, T-1, VOC)

    // workspace bump allocator (fp32 intermediates, ~4.2 MB total)
    float* base = (float*)d_ws;
    size_t off = 0;
    auto alloc = [&](size_t n) { float* p = base + off; off += n; return p; };
    float* vmean  = alloc((size_t)BB * FF);
    float* v_w    = alloc((size_t)BB * LL * 128);
    float* h1     = alloc((size_t)BB * HH);
    float* h2     = alloc((size_t)BB * HH);          // adjacent to h1
    float* x_att  = alloc((size_t)BB * ATT_IN);
    float* gi_a   = alloc((size_t)BB * G3H);
    float* gh_a   = alloc((size_t)BB * G3H);
    float* h_wbuf = alloc((size_t)BB * 128);
    float* attw   = alloc((size_t)BB * LL);
    float* mixbuf = alloc((size_t)BB * FF);
    float* x_lang = alloc((size_t)BB * LANG_IN);
    float* gi_l   = alloc((size_t)BB * G3H);
    float* gh_l   = alloc((size_t)BB * G3H);

    // ---- loop-invariant precompute ----
    vmean_kernel<<<(BB * FF + 255) / 256, 256, 0, stream>>>(fv, vmean);
    // v_w = feature_vector(2304,1024) @ fc1_w^T + fc1_b   -> (2304,128)
    launch_gemm(fv, FF, fc1_w, FF, fc1_b, v_w, 128, BB * LL, 128, FF, stream);
    zero_kernel<<<(2 * BB * HH + 255) / 256, 256, 0, stream>>>(h1, 2 * BB * HH);

    // ---- recurrence ----
    for (int t = 0; t < TSTEPS; ++t) {
        pack_att_kernel<<<(BB * ATT_IN + 255) / 256, 256, 0, stream>>>(h2, gt, vmean, x_att, t);
        // attention GRU
        launch_gemm(x_att, ATT_IN, att_w_ih, ATT_IN, att_b_ih, gi_a, G3H, BB, G3H, ATT_IN, stream);
        launch_gemm(h1, HH, att_w_hh, HH, att_b_hh, gh_a, G3H, BB, G3H, HH, stream);
        gru_gate_kernel<<<(BB * HH + 255) / 256, 256, 0, stream>>>(gi_a, gh_a, h1);
        // attention MLP + softmax + mixing
        launch_gemm(h1, HH, fc2_w, HH, fc2_b, h_wbuf, 128, BB, 128, HH, stream);
        att_score_kernel<<<BB, 64, 0, stream>>>(v_w, h_wbuf, fc3_w, fc3_b, attw);
        mix_kernel<<<(BB * FF + 255) / 256, 256, 0, stream>>>(attw, fv, mixbuf);
        // language GRU
        pack_lang_kernel<<<(BB * LANG_IN + 255) / 256, 256, 0, stream>>>(mixbuf, h1, x_lang);
        launch_gemm(x_lang, LANG_IN, lang_w_ih, LANG_IN, lang_b_ih, gi_l, G3H, BB, G3H, LANG_IN, stream);
        launch_gemm(h2, HH, lang_w_hh, HH, lang_b_hh, gh_l, G3H, BB, G3H, HH, stream);
        gru_gate_kernel<<<(BB * HH + 255) / 256, 256, 0, stream>>>(gi_l, gh_l, h2);
        // vocab projection straight into d_out: row b, col v at out[b*(20*VV) + t*VV + v]
        launch_gemm(h2, HH, dense_w, HH, dense_b, out + (size_t)t * VV, TSTEPS * VV,
                    BB, VV, HH, stream);
    }
}
